// NunchakuSDXLBasicTransformerBlock_3186865734264
// MI455X (gfx1250) — compile-verified, matrix operand order checked by reading, not run on hardware
//
#include <hip/hip_runtime.h>
#include <stdint.h>

// ---------------------------------------------------------------------------
// Types / helpers
// ---------------------------------------------------------------------------
typedef __attribute__((ext_vector_type(16))) _Float16 v16h;
typedef __attribute__((ext_vector_type(8)))  float    v8f;

union Frag16 { v16h h; uint4 u[2]; };

static __device__ __forceinline__ v16h ld_frag16(const _Float16* p) {
  // 8 contiguous halves at p, next 8 at p+16 (CDNA5 16-bit A/B operand layout)
  Frag16 f;
  f.u[0] = *(const uint4*)(p);
  f.u[1] = *(const uint4*)(p + 16);
  return f.h;
}

#define WMMA_F32_16x16x32_F16(A, B, C) \
  __builtin_amdgcn_wmma_f32_16x16x32_f16(false, (A), false, (B), (short)0, (C), false, false)

// Async global -> LDS copy, 16B per lane, tracked by ASYNCcnt (CDNA5 §15.18.3).
// GVS mode: mem_addr = SGPR64 base + per-lane signed 32-bit byte offset.
static __device__ __forceinline__ void async_ld_b128(unsigned lds_byte_addr,
                                                     unsigned gl_byte_off,
                                                     unsigned long long base) {
  asm volatile("global_load_async_to_lds_b128 %0, %1, %2"
               :: "v"(lds_byte_addr), "v"(gl_byte_off), "s"(base)
               : "memory");
}
#define WAIT_ASYNCCNT(n) asm volatile("s_wait_asynccnt " #n ::: "memory")

static __device__ __forceinline__ unsigned lds_addr32(const void* p) {
  // flat LDS aperture: addr[31:0] is the LDS byte offset
  return (unsigned)(uintptr_t)p;
}

// ---------------------------------------------------------------------------
// Group-64 symmetric int4 fake-quant (or plain convert) fp32 -> f16
// one wave per group of 64 contiguous elements
// ---------------------------------------------------------------------------
__global__ __launch_bounds__(256) void fq_groups_kernel(
    const float* __restrict__ in, _Float16* __restrict__ out,
    long long ngroups, int quantize) {
  long long g = (long long)blockIdx.x * 8 + (threadIdx.x >> 5);
  if (g >= ngroups) return;
  int lid = threadIdx.x & 31;
  long long base = g * 64;
  float a = in[base + lid];
  float b = in[base + lid + 32];
  if (quantize) {
    float m = fmaxf(fabsf(a), fabsf(b));
#pragma unroll
    for (int d = 16; d >= 1; d >>= 1) m = fmaxf(m, __shfl_xor(m, d, 32));
    float sc = m * (1.0f / 7.0f) + 1e-8f;
    float rs = 1.0f / sc;
    a = fminf(fmaxf(rintf(a * rs), -8.f), 7.f) * sc;
    b = fminf(fmaxf(rintf(b * rs), -8.f), 7.f) * sc;
  }
  out[base + lid]      = (_Float16)a;
  out[base + lid + 32] = (_Float16)b;
}

// ---------------------------------------------------------------------------
// LayerNorm over last dim (D<=1280) + group-64 fake-quant; writes fp32 LN and
// f16 fq(LN). One 256-thread block per row.
// ---------------------------------------------------------------------------
__global__ __launch_bounds__(256) void ln_fq_kernel(
    const float* __restrict__ x, const float* __restrict__ w, const float* __restrict__ b,
    float* __restrict__ ln_out, _Float16* __restrict__ fq_out, int D) {
  int row = blockIdx.x;
  const float* xr = x + (size_t)row * D;
  __shared__ float lbuf[1280];
  __shared__ float red0[8], red1[8];
  float s = 0.f, ss = 0.f;
  for (int i = threadIdx.x; i < D; i += 256) { float v = xr[i]; s += v; ss += v * v; }
#pragma unroll
  for (int d = 16; d >= 1; d >>= 1) { s += __shfl_xor(s, d, 32); ss += __shfl_xor(ss, d, 32); }
  int wid = threadIdx.x >> 5, lid = threadIdx.x & 31;
  if (lid == 0) { red0[wid] = s; red1[wid] = ss; }
  __syncthreads();
  float tot = 0.f, tot2 = 0.f;
#pragma unroll
  for (int i = 0; i < 8; ++i) { tot += red0[i]; tot2 += red1[i]; }
  float mean = tot / (float)D;
  float var  = tot2 / (float)D - mean * mean;
  float inv  = rsqrtf(var + 1e-5f);
  for (int i = threadIdx.x; i < D; i += 256) {
    float v = (xr[i] - mean) * inv * w[i] + b[i];
    lbuf[i] = v;
    ln_out[(size_t)row * D + i] = v;
  }
  __syncthreads();
  int ngroups = D >> 6;
  for (int g = wid; g < ngroups; g += 8) {
    float a = lbuf[g * 64 + lid];
    float c = lbuf[g * 64 + 32 + lid];
    float m = fmaxf(fabsf(a), fabsf(c));
#pragma unroll
    for (int d = 16; d >= 1; d >>= 1) m = fmaxf(m, __shfl_xor(m, d, 32));
    float sc = m * (1.0f / 7.0f) + 1e-8f;
    float rs = 1.0f / sc;
    a = fminf(fmaxf(rintf(a * rs), -8.f), 7.f) * sc;
    c = fminf(fmaxf(rintf(c * rs), -8.f), 7.f) * sc;
    fq_out[(size_t)row * D + g * 64 + lid]      = (_Float16)a;
    fq_out[(size_t)row * D + g * 64 + 32 + lid] = (_Float16)c;
  }
}

// ---------------------------------------------------------------------------
// WMMA GEMM: C[M,N] = A[M,K] * B[N,K]^T  (A,B f16 row-major, C fp32)
// Workgroup tile 128x128, BK=32, 8 waves (2x4), 8 WMMA tiles per wave.
// Global->LDS staging uses global_load_async_to_lds_b128 with double-buffered
// LDS tiles; the next K-step's copies overlap the current step's WMMAs.
// ---------------------------------------------------------------------------
__global__ __launch_bounds__(256) void wmma_gemm_kernel(
    const _Float16* __restrict__ A, const _Float16* __restrict__ B,
    float* __restrict__ C, int M, int N, int K) {
  __shared__ _Float16 As[2][128 * 40];
  __shared__ _Float16 Bs[2][128 * 40];
  const unsigned BUFSZ = 128 * 40 * 2;  // bytes per buffer

  int tid  = threadIdx.x;
  int wid  = tid >> 5;
  int lane = tid & 31;
  int wm = wid >> 2;       // 0..1 -> 64 rows each
  int wn = wid & 3;        // 0..3 -> 32 cols each
  int half = lane >> 4;    // 0/1 : K-interleave select
  int l16  = lane & 15;
  int rowBase = blockIdx.y * 128;
  int colBase = blockIdx.x * 128;

  // Two 16B chunks per thread per matrix: chunk c -> row c>>2, k-sub c&3.
  int c0 = tid, c1 = tid + 256;
  int r0 = c0 >> 2, kc0 = c0 & 3;
  int r1 = c1 >> 2, kc1 = c1 & 3;
  int ga0 = rowBase + r0; ga0 = ga0 < M ? ga0 : M - 1;
  int ga1 = rowBase + r1; ga1 = ga1 < M ? ga1 : M - 1;
  int gb0 = colBase + r0; gb0 = gb0 < N ? gb0 : N - 1;
  int gb1 = colBase + r1; gb1 = gb1 < N ? gb1 : N - 1;
  // per-lane byte offsets (K-step folded into the uniform SGPR base)
  unsigned aoff0 = (unsigned)(((size_t)ga0 * K + kc0 * 8) * 2);
  unsigned aoff1 = (unsigned)(((size_t)ga1 * K + kc1 * 8) * 2);
  unsigned boff0 = (unsigned)(((size_t)gb0 * K + kc0 * 8) * 2);
  unsigned boff1 = (unsigned)(((size_t)gb1 * K + kc1 * 8) * 2);
  unsigned lA0 = lds_addr32(&As[0][r0 * 40 + kc0 * 8]);
  unsigned lA1 = lds_addr32(&As[0][r1 * 40 + kc1 * 8]);
  unsigned lB0 = lds_addr32(&Bs[0][r0 * 40 + kc0 * 8]);
  unsigned lB1 = lds_addr32(&Bs[0][r1 * 40 + kc1 * 8]);

  auto issue_tile = [&](int k0, int buf) {
    unsigned long long ab = (unsigned long long)(uintptr_t)A + (unsigned long long)k0 * 2;
    unsigned long long bb = (unsigned long long)(uintptr_t)B + (unsigned long long)k0 * 2;
    unsigned boffset = buf ? BUFSZ : 0u;
    async_ld_b128(lA0 + boffset, aoff0, ab);
    async_ld_b128(lA1 + boffset, aoff1, ab);
    async_ld_b128(lB0 + boffset, boff0, bb);
    async_ld_b128(lB1 + boffset, boff1, bb);
  };

  v8f acc[4][2] = {};

  int nk = K >> 5;
  issue_tile(0, 0);                    // prologue: 4 async ops outstanding
  for (int kt = 0; kt < nk; ++kt) {
    int cur = kt & 1;
    if (kt + 1 < nk) {
      issue_tile((kt + 1) << 5, cur ^ 1);   // 8 outstanding
      WAIT_ASYNCCNT(4);                     // tile kt landed in LDS
    } else {
      WAIT_ASYNCCNT(0);
    }
    __syncthreads();                        // all waves' tile-kt data visible

    v16h af[4], bf[2];
#pragma unroll
    for (int i = 0; i < 4; ++i)
      af[i] = ld_frag16(&As[cur][(wm * 64 + i * 16 + l16) * 40 + half * 8]);
#pragma unroll
    for (int j = 0; j < 2; ++j)
      bf[j] = ld_frag16(&Bs[cur][(wn * 32 + j * 16 + l16) * 40 + half * 8]);

#pragma unroll
    for (int i = 0; i < 4; ++i)
#pragma unroll
      for (int j = 0; j < 2; ++j)
        acc[i][j] = WMMA_F32_16x16x32_F16(af[i], bf[j], acc[i][j]);

    __syncthreads();                        // done reading buf cur before reuse
  }

  // Store: VGPR r -> row (r + half*8), col = l16 within each 16x16 tile.
#pragma unroll
  for (int i = 0; i < 4; ++i)
#pragma unroll
    for (int j = 0; j < 2; ++j)
#pragma unroll
      for (int r = 0; r < 8; ++r) {
        int row = rowBase + wm * 64 + i * 16 + half * 8 + r;
        int col = colBase + wn * 32 + j * 16 + l16;
        if (row < M && col < N) C[(size_t)row * N + col] = acc[i][j][r];
      }
}

// ---------------------------------------------------------------------------
// Low-rank down-projection: t[M,32] = A[M,K] @ Ld[K,32]   (fp32, tiny FLOPs)
// ---------------------------------------------------------------------------
__global__ __launch_bounds__(32) void lr_down_kernel(
    const float* __restrict__ A, const float* __restrict__ Ld,
    float* __restrict__ t, int K) {
  int m = blockIdx.x;
  int r = threadIdx.x;            // 0..31
  const float* ar = A + (size_t)m * K;
  float acc = 0.f;
  for (int k = 0; k < K; ++k) acc = fmaf(ar[k], Ld[(size_t)k * 32 + r], acc);
  t[(size_t)m * 32 + r] = acc;
}

// ---------------------------------------------------------------------------
// y[M,N] += t[M,32] @ Lu[32,N] (+ bias)
// ---------------------------------------------------------------------------
__global__ __launch_bounds__(256) void lr_up_bias_kernel(
    float* __restrict__ y, const float* __restrict__ t,
    const float* __restrict__ Lu, const float* __restrict__ bias,
    long long M, int N) {
  long long idx = (long long)blockIdx.x * 256 + threadIdx.x;
  long long total = M * (long long)N;
  if (idx >= total) return;
  long long m = idx / N;
  int n = (int)(idx - m * N);
  const float* tm = t + m * 32;
  float acc = bias ? bias[n] : 0.f;
#pragma unroll
  for (int r = 0; r < 32; ++r) acc = fmaf(tm[r], Lu[(size_t)r * N + n], acc);
  y[idx] += acc;
}

// ---------------------------------------------------------------------------
// Flash attention (dh=64, 20 heads). Grid: (S/64, B*H); 4 waves, 16 Q rows/wave.
// QK^T and P@V both through v_wmma_f32_16x16x32_f16.
// ---------------------------------------------------------------------------
__global__ __launch_bounds__(128) void attn_kernel(
    const float* __restrict__ Q, int qstride, int qcoff,
    const float* __restrict__ Kp, int kstride, int kcoff,
    const float* __restrict__ Vp, int vstride, int vcoff,
    float* __restrict__ Out, int S, int T, int H, float scale) {
  const int DH = 64;
  int wid  = threadIdx.x >> 5;
  int lane = threadIdx.x & 31;
  int half = lane >> 4;
  int l16  = lane & 15;
  int b = blockIdx.y / H;
  int h = blockIdx.y % H;
  int s0 = blockIdx.x * 64 + wid * 16;

  __shared__ _Float16 Pl[4][16 * 40];

  // Q fragments (16 rows x 64 dh), scale folded in. Two K-chunks of 32.
  v16h aq[2];
  {
    const float* qrow = Q + ((size_t)(b * S + s0 + l16)) * qstride + qcoff + h * DH;
#pragma unroll
    for (int kw = 0; kw < 2; ++kw) {
      int kb = kw * 32 + half * 8;
#pragma unroll
      for (int j = 0; j < 8; ++j) aq[kw][j]     = (_Float16)(qrow[kb + j] * scale);
#pragma unroll
      for (int j = 0; j < 8; ++j) aq[kw][8 + j] = (_Float16)(qrow[kb + 16 + j] * scale);
    }
  }

  float mi[8], li[8];
  v8f o[4] = {};
#pragma unroll
  for (int r = 0; r < 8; ++r) { mi[r] = -1e30f; li[r] = 0.f; }

  for (int t0 = 0; t0 < T; t0 += 32) {
    // ---- S = Q K^T for 32 keys (two 16-col tiles) ----
    v8f sacc[2];
#pragma unroll
    for (int ct = 0; ct < 2; ++ct) {
      int key = t0 + ct * 16 + l16;
      int kk  = key < T ? key : T - 1;
      const float* krow = Kp + ((size_t)(b * T + kk)) * kstride + kcoff + h * DH;
      __builtin_prefetch(krow + 32 * kstride, 0, 0);
      v8f sv = {};
#pragma unroll
      for (int kw = 0; kw < 2; ++kw) {
        v16h bk;
        int kb = kw * 32 + half * 8;
#pragma unroll
        for (int j = 0; j < 8; ++j) bk[j]     = (_Float16)krow[kb + j];
#pragma unroll
        for (int j = 0; j < 8; ++j) bk[8 + j] = (_Float16)krow[kb + 16 + j];
        sv = WMMA_F32_16x16x32_F16(aq[kw], bk, sv);
      }
      if (key >= T) {
#pragma unroll
        for (int r = 0; r < 8; ++r) sv[r] = -1e30f;
      }
      sacc[ct] = sv;
    }

    // ---- online softmax stats (row r lives in vgpr r across a 16-lane group) ----
    float cmax[8];
#pragma unroll
    for (int r = 0; r < 8; ++r) cmax[r] = fmaxf(sacc[0][r], sacc[1][r]);
#pragma unroll
    for (int d = 8; d >= 1; d >>= 1)
#pragma unroll
      for (int r = 0; r < 8; ++r) cmax[r] = fmaxf(cmax[r], __shfl_xor(cmax[r], d, 32));

    float corr[8];
#pragma unroll
    for (int r = 0; r < 8; ++r) {
      float mn = fmaxf(mi[r], cmax[r]);
      corr[r] = __expf(mi[r] - mn);
      mi[r] = mn;
    }
#pragma unroll
    for (int ct = 0; ct < 2; ++ct)
#pragma unroll
      for (int r = 0; r < 8; ++r) sacc[ct][r] = __expf(sacc[ct][r] - mi[r]);

    float psum[8];
#pragma unroll
    for (int r = 0; r < 8; ++r) psum[r] = sacc[0][r] + sacc[1][r];
#pragma unroll
    for (int d = 8; d >= 1; d >>= 1)
#pragma unroll
      for (int r = 0; r < 8; ++r) psum[r] += __shfl_xor(psum[r], d, 32);
#pragma unroll
    for (int r = 0; r < 8; ++r) li[r] = li[r] * corr[r] + psum[r];
#pragma unroll
    for (int n = 0; n < 4; ++n)
#pragma unroll
      for (int r = 0; r < 8; ++r) o[n][r] *= corr[r];

    // ---- relayout P (16x32) to canonical A fragment via per-wave LDS ----
    int prow = half * 8;
#pragma unroll
    for (int ct = 0; ct < 2; ++ct)
#pragma unroll
      for (int r = 0; r < 8; ++r)
        Pl[wid][(prow + r) * 40 + ct * 16 + l16] = (_Float16)sacc[ct][r];
    v16h ap = ld_frag16(&Pl[wid][l16 * 40 + half * 8]);

    // ---- O += P @ V (4 dh-chunks of 16 cols) ----
#pragma unroll
    for (int n = 0; n < 4; ++n) {
      v16h bv;
      int kb = half * 8;
#pragma unroll
      for (int j = 0; j < 8; ++j) {
        int key = t0 + kb + j; key = key < T ? key : T - 1;
        bv[j] = (_Float16)Vp[((size_t)(b * T + key)) * vstride + vcoff + h * DH + n * 16 + l16];
      }
#pragma unroll
      for (int j = 0; j < 8; ++j) {
        int key = t0 + kb + 16 + j; key = key < T ? key : T - 1;
        bv[8 + j] = (_Float16)Vp[((size_t)(b * T + key)) * vstride + vcoff + h * DH + n * 16 + l16];
      }
      o[n] = WMMA_F32_16x16x32_F16(ap, bv, o[n]);
    }
  }

  // ---- normalize + store (b, s, h*64+d) ----
#pragma unroll
  for (int n = 0; n < 4; ++n)
#pragma unroll
    for (int r = 0; r < 8; ++r) {
      int row = s0 + half * 8 + r;
      float val = o[n][r] / li[r];
      Out[((size_t)(b * S + row)) * (H * DH) + h * DH + n * 16 + l16] = val;
    }
}

// ---------------------------------------------------------------------------
// GEGLU activation: h = a * gelu_tanh(gate), g = [a | gate] (M x 2*inner)
// ---------------------------------------------------------------------------
__global__ __launch_bounds__(256) void geglu_kernel(
    const float* __restrict__ g, float* __restrict__ h, long long M, int inner) {
  long long idx = (long long)blockIdx.x * 256 + threadIdx.x;
  if (idx >= M * (long long)inner) return;
  long long m = idx / inner;
  int i = (int)(idx - m * inner);
  float a  = g[m * (2LL * inner) + i];
  float xg = g[m * (2LL * inner) + inner + i];
  float t  = 0.7978845608028654f * (xg + 0.044715f * xg * xg * xg);
  h[idx] = a * 0.5f * xg * (1.f + tanhf(t));
}

__global__ __launch_bounds__(256) void add_kernel(
    const float* __restrict__ a, const float* __restrict__ b,
    float* __restrict__ o, long long n) {
  long long i = (long long)blockIdx.x * 256 + threadIdx.x;
  if (i < n) o[i] = a[i] + b[i];
}

// ---------------------------------------------------------------------------
// Host-side orchestration
// ---------------------------------------------------------------------------
extern "C" void kernel_launch(void* const* d_in, const int* in_sizes, int n_in,
                              void* d_out, int out_size, void* d_ws, size_t ws_size,
                              hipStream_t stream) {
  (void)in_sizes; (void)n_in; (void)out_size; (void)ws_size;

  const float* x      = (const float*)d_in[0];
  const float* ctx    = (const float*)d_in[1];
  const float* n1w    = (const float*)d_in[2];
  const float* n1b    = (const float*)d_in[3];
  const float* n2w    = (const float*)d_in[4];
  const float* n2b    = (const float*)d_in[5];
  const float* n3w    = (const float*)d_in[6];
  const float* n3b    = (const float*)d_in[7];
  const float* qkvW   = (const float*)d_in[8];
  const float* qkvLd  = (const float*)d_in[9];
  const float* qkvLu  = (const float*)d_in[10];
  const float* o1W    = (const float*)d_in[11];
  const float* o1Ld   = (const float*)d_in[12];
  const float* o1Lu   = (const float*)d_in[13];
  const float* o1b    = (const float*)d_in[14];
  const float* qW     = (const float*)d_in[15];
  const float* qLd    = (const float*)d_in[16];
  const float* qLu    = (const float*)d_in[17];
  const float* kW     = (const float*)d_in[18];
  const float* vW     = (const float*)d_in[19];
  const float* o2W    = (const float*)d_in[20];
  const float* o2Ld   = (const float*)d_in[21];
  const float* o2Lu   = (const float*)d_in[22];
  const float* o2b    = (const float*)d_in[23];
  const float* ggW    = (const float*)d_in[24];
  const float* ggLd   = (const float*)d_in[25];
  const float* ggLu   = (const float*)d_in[26];
  const float* ggb    = (const float*)d_in[27];
  const float* ffW    = (const float*)d_in[28];
  const float* ffLd   = (const float*)d_in[29];
  const float* ffLu   = (const float*)d_in[30];
  const float* ffb    = (const float*)d_in[31];
  float* out = (float*)d_out;

  const int B = 4, S = 1024, D = 1280, Cd = 2048, INNER = 5120, N3 = 3840;
  const int H = 20, Tc = 77;
  const int M = B * S;            // 4096
  const int CTXR = B * Tc;        // 308
  const float scale = 0.125f;     // 1/sqrt(64)

  char* ws = (char*)d_ws;
  size_t off = 0;
  auto alloc = [&](size_t bytes) -> void* {
    off = (off + 255) & ~(size_t)255;
    void* p = ws + off;
    off += bytes;
    return p;
  };

  // f16 weight / activation buffers
  _Float16* wQKV = (_Float16*)alloc((size_t)N3 * D * 2);
  _Float16* wO1  = (_Float16*)alloc((size_t)D * D * 2);
  _Float16* wQ   = (_Float16*)alloc((size_t)D * D * 2);
  _Float16* wK   = (_Float16*)alloc((size_t)D * Cd * 2);
  _Float16* wV   = (_Float16*)alloc((size_t)D * Cd * 2);
  _Float16* wO2  = (_Float16*)alloc((size_t)D * D * 2);
  _Float16* wGG  = (_Float16*)alloc((size_t)2 * INNER * D * 2);
  _Float16* wFF  = (_Float16*)alloc((size_t)D * INNER * 2);
  _Float16* ctx16 = (_Float16*)alloc((size_t)CTXR * Cd * 2);
  _Float16* act16 = (_Float16*)alloc((size_t)M * INNER * 2);   // generic fq activations
  // fp32 scratch
  float* ln32   = (float*)alloc((size_t)M * D * 4);
  float* t32    = (float*)alloc((size_t)M * 32 * 4);
  float* gem32  = (float*)alloc((size_t)M * N3 * 4);           // qkv / projection outputs
  float* attn32 = (float*)alloc((size_t)M * D * 4);
  float* xbuf   = (float*)alloc((size_t)M * D * 4);
  float* kc32   = (float*)alloc((size_t)CTXR * D * 4);
  float* vc32   = (float*)alloc((size_t)CTXR * D * 4);
  float* big32  = (float*)alloc((size_t)M * 2 * INNER * 4);    // geglu out
  float* h32    = (float*)alloc((size_t)M * INNER * 4);

  auto fq = [&](const float* in, _Float16* o, long long elems, int quant) {
    long long groups = elems >> 6;
    unsigned blocks = (unsigned)((groups + 7) / 8);
    fq_groups_kernel<<<blocks, 256, 0, stream>>>(in, o, groups, quant);
  };
  auto gemm = [&](const _Float16* A, const _Float16* Bw, float* Cp, int M_, int N_, int K_) {
    dim3 g((N_ + 127) / 128, (M_ + 127) / 128);
    wmma_gemm_kernel<<<g, 256, 0, stream>>>(A, Bw, Cp, M_, N_, K_);
  };
  auto lrdown = [&](const float* A, const float* Ld, int K_) {
    lr_down_kernel<<<M, 32, 0, stream>>>(A, Ld, t32, K_);
  };
  auto lrup = [&](float* y, const float* Lu, const float* bias, int N_) {
    long long total = (long long)M * N_;
    unsigned blocks = (unsigned)((total + 255) / 256);
    lr_up_bias_kernel<<<blocks, 256, 0, stream>>>(y, t32, Lu, bias, (long long)M, N_);
  };
  auto resadd = [&](const float* a, const float* b, float* o) {
    long long n = (long long)M * D;
    add_kernel<<<(unsigned)((n + 255) / 256), 256, 0, stream>>>(a, b, o, n);
  };

  // ---- one-time per launch: quantize/convert weights + context to f16 ----
  fq(qkvW, wQKV, (long long)N3 * D, 1);
  fq(o1W,  wO1,  (long long)D * D, 1);
  fq(qW,   wQ,   (long long)D * D, 1);
  fq(kW,   wK,   (long long)D * Cd, 0);   // plain fp path in reference
  fq(vW,   wV,   (long long)D * Cd, 0);
  fq(o2W,  wO2,  (long long)D * D, 1);
  fq(ggW,  wGG,  (long long)2 * INNER * D, 1);
  fq(ffW,  wFF,  (long long)D * INNER, 1);
  fq(ctx,  ctx16, (long long)CTXR * Cd, 0);

  // ================= stage 1: self-attention =================
  ln_fq_kernel<<<M, 256, 0, stream>>>(x, n1w, n1b, ln32, act16, D);
  lrdown(ln32, qkvLd, D);
  gemm(act16, wQKV, gem32, M, N3, D);
  lrup(gem32, qkvLu, nullptr, N3);
  attn_kernel<<<dim3(S / 64, B * H), 128, 0, stream>>>(
      gem32, N3, 0, gem32, N3, D, gem32, N3, 2 * D, attn32, S, S, H, scale);
  fq(attn32, act16, (long long)M * D, 1);
  lrdown(attn32, o1Ld, D);
  gemm(act16, wO1, gem32, M, D, D);
  lrup(gem32, o1Lu, o1b, D);
  resadd(x, gem32, xbuf);

  // ================= stage 2: cross-attention =================
  ln_fq_kernel<<<M, 256, 0, stream>>>(xbuf, n2w, n2b, ln32, act16, D);
  lrdown(ln32, qLd, D);
  gemm(act16, wQ, gem32, M, D, D);
  lrup(gem32, qLu, nullptr, D);
  gemm(ctx16, wK, kc32, CTXR, D, Cd);
  gemm(ctx16, wV, vc32, CTXR, D, Cd);
  attn_kernel<<<dim3(S / 64, B * H), 128, 0, stream>>>(
      gem32, D, 0, kc32, D, 0, vc32, D, 0, attn32, S, Tc, H, scale);
  fq(attn32, act16, (long long)M * D, 1);
  lrdown(attn32, o2Ld, D);
  gemm(act16, wO2, gem32, M, D, D);
  lrup(gem32, o2Lu, o2b, D);
  resadd(xbuf, gem32, xbuf);

  // ================= stage 3: GEGLU feed-forward =================
  ln_fq_kernel<<<M, 256, 0, stream>>>(xbuf, n3w, n3b, ln32, act16, D);
  lrdown(ln32, ggLd, D);
  gemm(act16, wGG, big32, M, 2 * INNER, D);
  lrup(big32, ggLu, ggb, 2 * INNER);
  {
    long long n = (long long)M * INNER;
    geglu_kernel<<<(unsigned)((n + 255) / 256), 256, 0, stream>>>(big32, h32, (long long)M, INNER);
  }
  fq(h32, act16, (long long)M * INNER, 1);
  lrdown(h32, ffLd, INNER);
  gemm(act16, wFF, gem32, M, D, INNER);
  lrup(gem32, ffLu, ffb, D);
  resadd(xbuf, gem32, out);
}
